// match_features_fst_17712445128929
// MI455X (gfx1250) — matrix operand order repaired and checked by reading, not compiled, not run on hardware
//
#include <hip/hip_runtime.h>

typedef __attribute__((ext_vector_type(16))) _Float16 v16h;
typedef __attribute__((ext_vector_type(8)))  float    v8f;

namespace {
constexpr int Bn = 2, Vn = 2, Cn = 64, Dn = 48, Hn = 96, Wn = 128;
constexpr int TILE = 16;
constexpr int HW   = Hn * Wn;
}

__global__ __launch_bounds__(32)
void cost_volume_wmma(const float* __restrict__ depth_bins,   // [B,D]
                      const float* __restrict__ ref_feat,     // [B,C,H,W]
                      const float* __restrict__ lookup,       // [B,V,C,H,W]
                      const float* __restrict__ poses,        // [B,V,4,4]
                      const float* __restrict__ Ks,           // [B,V,4,4]
                      const float* __restrict__ invK,         // [B,4,4]
                      float* __restrict__ out_cost,           // [B,D,H,W]
                      float* __restrict__ out_mask)           // [B,D,H,W]
{
  const int l    = threadIdx.x;
  const int p    = l & 15;    // pixel within tile (both half-lanes share it)
  const int half = l >> 4;    // which channel-half of the fragments this lane owns

  const int ntx = Wn / TILE;
  const int bx  = blockIdx.x;
  const int xt  = (bx % ntx) * TILE;
  const int y   = (bx / ntx) % Hn;
  const int b   = bx / (ntx * Hn);
  const int x   = xt + p;

  // ---- A fragments (A = ref^T, 16x32 per K-chunk), loaded straight from
  // global into this kernel's contiguous-k layout: lane l holds M = l%16, element e holds
  // K = e + 16*(l/16)  (+32 for the second chunk).
  v16h a0, a1;
  {
    const float* rp = ref_feat + (size_t)b * Cn * HW + (size_t)y * Wn + x
                               + (size_t)(16 * half) * HW;
    #pragma unroll
    for (int e = 0; e < 8; ++e) {
      a0[e]     = (_Float16)rp[(size_t)(e     ) * HW];
      a0[e + 8] = (_Float16)rp[(size_t)(e + 8) * HW];
      a1[e]     = (_Float16)rp[(size_t)(e + 32) * HW];
      a1[e + 8] = (_Float16)rp[(size_t)(e + 40) * HW];
    }
  }

  // ---- per-view projection rows: M = Ks * pose * invK (rows 0..2), folded
  // with this lane's pixel: proj = dep * r + t
  float rx[Vn], ry[Vn], rz[Vn], tx[Vn], ty[Vn], tz[Vn];
  #pragma unroll
  for (int v = 0; v < Vn; ++v) {
    const float* Kb = Ks    + (size_t)(b * Vn + v) * 16;
    const float* Tb = poses + (size_t)(b * Vn + v) * 16;
    const float* Ib = invK  + (size_t)b * 16;
    float TI[16];
    #pragma unroll
    for (int i = 0; i < 4; ++i)
      #pragma unroll
      for (int j = 0; j < 4; ++j) {
        float s = 0.f;
        #pragma unroll
        for (int k = 0; k < 4; ++k) s += Tb[i*4+k] * Ib[k*4+j];
        TI[i*4+j] = s;
      }
    float M[3][4];
    #pragma unroll
    for (int i = 0; i < 3; ++i)
      #pragma unroll
      for (int j = 0; j < 4; ++j) {
        float s = 0.f;
        #pragma unroll
        for (int k = 0; k < 4; ++k) s += Kb[i*4+k] * TI[k*4+j];
        M[i][j] = s;
      }
    const float fx = (float)x, fy = (float)y;
    rx[v] = M[0][0]*fx + M[0][1]*fy + M[0][2]; tx[v] = M[0][3];
    ry[v] = M[1][0]*fx + M[1][1]*fy + M[1][2]; ty[v] = M[1][3];
    rz[v] = M[2][0]*fx + M[2][1]*fy + M[2][2]; tz[v] = M[2][3];
  }

  const float scale   = 1.0f / Cn;
  const float inv_hw  = 1.0f / (Wn * 0.5f);   // power of two: exact
  const float inv_hh  = 1.0f / (Hn * 0.5f);   // 1/48: 1-ulp vs divide, fine here
  const float* lkb    = lookup + (size_t)b * Vn * Cn * HW;

  for (int d = 0; d < Dn; ++d) {
    const float dep = depth_bins[b * Dn + d];
    v8f acc = {};          // accumulates both views' cost tile in f32
    float cnt = 0.f;       // per-lane validity count for pixel p

    #pragma unroll
    for (int v = 0; v < Vn; ++v) {
      // project pixel (x,y) at depth dep into view v (reference math; HW rcp
      // + reciprocal-constant normalization instead of IEEE divide chains)
      const float pxn = dep * rx[v] + tx[v];
      const float pyn = dep * ry[v] + ty[v];
      const float pzn = dep * rz[v] + tz[v] + 1e-8f;
      const float ipz = __builtin_amdgcn_rcpf(pzn);
      const float gx  = (pxn * ipz - Wn * 0.5f) * inv_hw;
      const float gy  = (pyn * ipz - Hn * 0.5f) * inv_hh;
      const bool ok   = !((gx > 1.f) || (gx < -1.f) || (gy > 1.f) || (gy < -1.f));
      const float vf  = ok ? 1.f : 0.f;   // validity factor (branchless)
      cnt += vf;

      // align_corners=True bilinear. Coordinates are clamped unconditionally
      // (safe addresses even for invalid/behind-camera pixels); the validity
      // factor is folded into wy0/wy1 which scales all 4 corner weights, so
      // invalid pixels contribute exactly-zero B columns (== reference's
      // cost*valid). No divergence: EXEC stays all-ones through the WMMAs.
      float xs = (gx + 1.f) * 0.5f * (Wn - 1);
      float ys = (gy + 1.f) * 0.5f * (Hn - 1);
      xs = fminf(fmaxf(xs, 0.f), (float)(Wn - 1));
      ys = fminf(fmaxf(ys, 0.f), (float)(Hn - 1));
      const float x0f = floorf(xs), y0f = floorf(ys);
      const float wx1 = xs - x0f,  wx0 = 1.f - wx1;
      const float wy1 = (ys - y0f) * vf, wy0 = (1.f - (ys - y0f)) * vf;
      const int xi0 = (int)x0f, yi0 = (int)y0f;
      const int xi1 = (xi0 + 1 > Wn - 1) ? Wn - 1 : xi0 + 1;  // weight 0 when clamped
      const int yi1 = (yi0 + 1 > Hn - 1) ? Hn - 1 : yi0 + 1;
      const int q00 = yi0 * Wn + xi0, q01 = yi0 * Wn + xi1;
      const int q10 = yi1 * Wn + xi0, q11 = yi1 * Wn + xi1;
      const float w00 = wx0*wy0, w01 = wx1*wy0, w10 = wx0*wy1, w11 = wx1*wy1;

      // four corner base pointers; the per-channel stride HW*4 = 49152 B is a
      // compile-time constant that fits the 24-bit global-load IOFFSET, so
      // every load below is base-reg + literal offset (max 47*49152B).
      const float* base = lkb + (size_t)v * Cn * HW + (size_t)(16 * half) * HW;
      const float* c00 = base + q00;
      const float* c01 = base + q01;
      const float* c10 = base + q10;
      const float* c11 = base + q11;

      // B fragments built directly in registers: lane l holds N = l%16,
      // element e holds K = e + 16*(l/16) (+32 for the second chunk) -- which
      // is exactly the set of channels this lane gathers for its own pixel.
      v16h b0, b1;
      #pragma unroll
      for (int e = 0; e < 16; ++e) {
        const size_t o0 = (size_t)e * HW;          // channels 16*half + e
        b0[e] = (_Float16)(c00[o0]*w00 + c01[o0]*w01 + c10[o0]*w10 + c11[o0]*w11);
        const size_t o1 = (size_t)(e + 32) * HW;   // channels 32 + 16*half + e
        b1[e] = (_Float16)(c00[o1]*w00 + c01[o1]*w01 + c10[o1]*w10 + c11[o1]*w11);
      }

      acc = __builtin_amdgcn_wmma_f32_16x16x32_f16(false, a0, false, b0,
                                                   (short)0, acc, false, false);
      acc = __builtin_amdgcn_wmma_f32_16x16x32_f16(false, a1, false, b1,
                                                   (short)0, acc, false, false);
    }

    // diag(R^T * W) = cost: lanes 0..7 hold pixels 0..7 in acc[lane],
    // lanes 24..31 hold pixels 8..15 in acc[lane-24]. These lanes' l%16
    // equals their pixel, so `cnt` is already the right validity count.
    if (l < 8 || l >= 24) {
      union { v8f v; float f[8]; } u; u.v = acc;
      const float dot2 = u.f[l & 7] * scale;
      const float c    = cnt;
      const float L1   = dot2 / (c + 1e-7f);
      const float mm   = (c == 0.f) ? 1.f : 0.f;
      const size_t o = (((size_t)b * Dn + d) * Hn + y) * Wn + xt + p;
      out_cost[o] = L1;
      out_mask[o] = mm;
    }
  }
}

extern "C" void kernel_launch(void* const* d_in, const int* in_sizes, int n_in,
                              void* d_out, int out_size, void* d_ws, size_t ws_size,
                              hipStream_t stream) {
  const float* depth_bins = (const float*)d_in[0];
  const float* ref_feat   = (const float*)d_in[1];
  const float* lookup     = (const float*)d_in[2];
  const float* poses      = (const float*)d_in[3];
  const float* Ks         = (const float*)d_in[4];
  const float* invKr      = (const float*)d_in[5];

  float* out_cost = (float*)d_out;
  float* out_mask = out_cost + (size_t)Bn * Dn * Hn * Wn;

  dim3 grid(Bn * Hn * (Wn / TILE));   // 1536 single-wave workgroups
  cost_volume_wmma<<<grid, 32, 0, stream>>>(depth_bins, ref_feat, lookup,
                                            poses, Ks, invKr,
                                            out_cost, out_mask);
}
